// Encoder_43696997269790
// MI455X (gfx1250) — matrix operand order repaired and checked by reading, not compiled, not run on hardware
//
#include <hip/hip_runtime.h>

// Problem constants from the reference
#define BB 64
#define LL 512
#define DD 512
#define TT 5000
#define NLAYER 2

typedef _Float16 half_t;
typedef _Float16 __attribute__((ext_vector_type(16))) v16h;
typedef _Float16 __attribute__((ext_vector_type(8)))  v8h;
typedef float    __attribute__((ext_vector_type(8)))  v8f;
typedef int      __attribute__((ext_vector_type(4)))  int4v;

// Pointer types matching the async builtin's signature (from hipcc diagnostic:
// param0 = 'int __vector(4) __device__ *'  i.e. AS(1) int4*)
typedef __attribute__((address_space(1))) int4v* g_int4p;
typedef __attribute__((address_space(3))) int4v* l_int4p;

// --------------------------------------------------------------------------
// CDNA5 async global->LDS copy (GLOBAL_LOAD_ASYNC_TO_LDS_B128, ASYNCcnt),
// with a portable fallback (global_load_b128 -> VGPR -> ds_store_b128).
// --------------------------------------------------------------------------
#if __has_builtin(__builtin_amdgcn_global_load_async_to_lds_b128)
#define USE_ASYNC 1
#else
#define USE_ASYNC 0
#endif

__device__ __forceinline__ void async_wait0() {
#if __has_builtin(__builtin_amdgcn_s_wait_asynccnt)
    __builtin_amdgcn_s_wait_asynccnt(0);
#else
    asm volatile("s_wait_asynccnt 0" ::: "memory");
#endif
}

__device__ __forceinline__ void async_cp16(const half_t* g, half_t* l) {
#if USE_ASYNC
    __builtin_amdgcn_global_load_async_to_lds_b128(
        (g_int4p)(const int4v*)g,
        (l_int4p)(int4v*)l,
        0, 0);
#else
    (void)g; (void)l;
#endif
}

// --------------------------------------------------------------------------
// LDS tile geometry: 128 rows x 32 halves per stage, rows padded to 40 halves
// (80B = 20 banks stride => conflict-free b128 fragment reads).
// --------------------------------------------------------------------------
#define SROW 40

// ---------------------------------------------------------------------------
// Batched 512x512x512 f16 GEMM, f32 accumulate.  C[b] = op(A[b] @ B[b])
// A: row-major [512,512] f16, batch stride sA
// B: stored transposed [N,K] row-major f16, batch stride sB (0 => shared W)
// C: STORE_T ? transposed [N,M] (packed v8h stores) : row-major [M,N]
// Block 256 threads (8 waves, 2x4), block tile 128x128, wave tile 64x32,
// K staged through double-buffered LDS in steps of 32.
// Grid: (N/128, M/128, batch)
// ---------------------------------------------------------------------------
template <bool RELU, bool STORE_T>
__global__ __launch_bounds__(256)
void gemm_wmma_f16(const half_t* __restrict__ A, unsigned long long sA,
                   const half_t* __restrict__ B, unsigned long long sB,
                   half_t* __restrict__ C, unsigned long long sC) {
    __shared__ half_t sAbuf[2][128 * SROW];   // 10KB per buffer
    __shared__ half_t sBbuf[2][128 * SROW];   // total 40KB LDS

    const int b = blockIdx.z;
    const half_t* Ag = A + (size_t)b * sA + (size_t)blockIdx.y * 128 * 512;
    const half_t* Bg = B + (size_t)b * sB + (size_t)blockIdx.x * 128 * 512;
    half_t*       Cb = C + (size_t)b * sC;

    const int t    = threadIdx.x;
    const int lane = t & 31;
    const int w    = t >> 5;
    const int wm   = w >> 2;                 // 0..1
    const int wn   = w & 3;                  // 0..3
    const int rl0  = wm * 64;                // wave-local row base (+mt*16)
    const int cl0  = wn * 32;                // wave-local col base (+nt*16)

    // Staging map: 512 16B chunks per 128x32 tile; thread t handles chunks
    // {t, t+256}: row = chunk>>2, 16B-chunk-in-row = chunk&3.
    const int r0s = t >> 2;                  // 0..63
    const int r1s = r0s + 64;                // 64..127
    const int cs  = (t & 3) * 8;             // column offset in halves

    // Fragment read map (V_WMMA_F32_16X16X32_F16, wave32)
    const int fr = lane & 15;                // A: row-in-tile / B: col-in-tile
    const int hs = lane >> 4;                // half-wave select

    v8f acc[4][2];
#pragma unroll
    for (int mt = 0; mt < 4; ++mt)
#pragma unroll
        for (int nt = 0; nt < 2; ++nt) acc[mt][nt] = (v8f)(0.0f);

    auto stage_load = [&](int k0, v8h r[4]) {
        r[0] = *(const v8h*)(Ag + (size_t)r0s * 512 + k0 + cs);
        r[1] = *(const v8h*)(Ag + (size_t)r1s * 512 + k0 + cs);
        r[2] = *(const v8h*)(Bg + (size_t)r0s * 512 + k0 + cs);
        r[3] = *(const v8h*)(Bg + (size_t)r1s * 512 + k0 + cs);
    };
    auto stage_store = [&](int buf, v8h r[4]) {
        *(v8h*)&sAbuf[buf][r0s * SROW + cs] = r[0];
        *(v8h*)&sAbuf[buf][r1s * SROW + cs] = r[1];
        *(v8h*)&sBbuf[buf][r0s * SROW + cs] = r[2];
        *(v8h*)&sBbuf[buf][r1s * SROW + cs] = r[3];
    };
    auto stage_async = [&](int k0, int buf) {
        async_cp16(Ag + (size_t)r0s * 512 + k0 + cs, &sAbuf[buf][r0s * SROW + cs]);
        async_cp16(Ag + (size_t)r1s * 512 + k0 + cs, &sAbuf[buf][r1s * SROW + cs]);
        async_cp16(Bg + (size_t)r0s * 512 + k0 + cs, &sBbuf[buf][r0s * SROW + cs]);
        async_cp16(Bg + (size_t)r1s * 512 + k0 + cs, &sBbuf[buf][r1s * SROW + cs]);
    };
    auto compute = [&](int cur) {
        const half_t* sAc = &sAbuf[cur][0];
        const half_t* sBc = &sBbuf[cur][0];
        v16h af[4], bf[2];
#pragma unroll
        for (int mt = 0; mt < 4; ++mt) {
            const half_t* q = sAc + (rl0 + mt * 16 + fr) * SROW + hs * 8;
            union { v16h v; v8h h[2]; } u;
            u.h[0] = *(const v8h*)(q);        // K hs*8 + 0..7
            u.h[1] = *(const v8h*)(q + 16);   // K hs*8 + 16..23
            af[mt] = u.v;
        }
#pragma unroll
        for (int nt = 0; nt < 2; ++nt) {
            const half_t* q = sBc + (cl0 + nt * 16 + fr) * SROW + hs * 8;
            union { v16h v; v8h h[2]; } u;
            u.h[0] = *(const v8h*)(q);        // K hs*8 + 0..7
            u.h[1] = *(const v8h*)(q + 16);   // K hs*8 + 16..23
            bf[nt] = u.v;
        }
#pragma unroll
        for (int mt = 0; mt < 4; ++mt)
#pragma unroll
            for (int nt = 0; nt < 2; ++nt)
                acc[mt][nt] = __builtin_amdgcn_wmma_f32_16x16x32_f16(
                    false, af[mt], false, bf[nt], (short)0, acc[mt][nt], false, false);
    };

#if USE_ASYNC
    stage_async(0, 0);
    async_wait0();
    __syncthreads();
    for (int s = 0; s < 16; ++s) {
        const int cur = s & 1;
        if (s < 15) stage_async((s + 1) * 32, cur ^ 1);  // overlaps with WMMA
        compute(cur);
        if (s < 15) async_wait0();
        __syncthreads();
    }
#else
    {
        v8h r[4];
        stage_load(0, r);
        stage_store(0, r);
    }
    __syncthreads();
    for (int s = 0; s < 16; ++s) {
        const int cur = s & 1;
        v8h r[4];
        if (s < 15) stage_load((s + 1) * 32, r);         // in flight during WMMA
        compute(cur);
        if (s < 15) stage_store(cur ^ 1, r);
        __syncthreads();
    }
#endif

    // C fragment: VGPR r -> M = r + 8*(lane>>4), N = lane&15 (per 16x16 tile)
    const int row0 = blockIdx.y * 128 + rl0;
    const int col0 = blockIdx.x * 128 + cl0;
    const int nl = lane & 15;
    const int mh = (lane >> 4) * 8;
#pragma unroll
    for (int mt = 0; mt < 4; ++mt) {
#pragma unroll
        for (int nt = 0; nt < 2; ++nt) {
            const int rbase = row0 + mt * 16 + mh;
            const int c     = col0 + nt * 16 + nl;
            if (STORE_T) {
                v8h outv;
#pragma unroll
                for (int r = 0; r < 8; ++r) {
                    float v = acc[mt][nt][r];
                    if (RELU) v = v > 0.0f ? v : 0.0f;
                    outv[r] = (half_t)v;
                }
                *(v8h*)(Cb + (size_t)c * 512 + rbase) = outv;   // contiguous 16B
            } else {
#pragma unroll
                for (int r = 0; r < 8; ++r) {
                    float v = acc[mt][nt][r];
                    if (RELU) v = v > 0.0f ? v : 0.0f;
                    Cb[(size_t)(rbase + r) * 512 + c] = (half_t)v;
                }
            }
        }
    }
}

// ---------------------------------------------------------------------------
// adj[b,i,j] = ui_adj[idx[b,i], idx[b,j]] + ui_adj[idx[b,j], idx[b,j]]  (f16)
// Grid: (L/16, B).  Column indices + diagonal staged in LDS, reused 16 rows.
// ---------------------------------------------------------------------------
__global__ __launch_bounds__(256)
void build_adj(const float* __restrict__ ui_adj, const int* __restrict__ event_type,
               half_t* __restrict__ adj) {
    __shared__ int   s_idx[LL];
    __shared__ float s_diag[LL];
    const int b = blockIdx.y, chunk = blockIdx.x, t = threadIdx.x;

    for (int j = t; j < LL; j += 256) {
        const int id = event_type[b * LL + j] - 1;
        s_idx[j]  = id;
        s_diag[j] = ui_adj[(size_t)id * TT + id];
    }
    __syncthreads();

    for (int r = 0; r < 16; ++r) {
        const int i = chunk * 16 + r;
        const size_t rowbase = (size_t)s_idx[i] * TT;
        half_t* dst = adj + ((size_t)b * LL + i) * LL;
        for (int j = t; j < LL; j += 256)
            dst[j] = (half_t)(ui_adj[rowbase + s_idx[j]] + s_diag[j]);
    }
}

// ---------------------------------------------------------------------------
// Transpose + f32->f16 convert: in [batch,R,C] f32 -> out [batch,C,R] f16.
// Block (32,8), grid (C/32, R/32, batch).
// ---------------------------------------------------------------------------
__global__ __launch_bounds__(256)
void transpose_cvt(const float* __restrict__ in, half_t* __restrict__ out,
                   int R, int C) {
    __shared__ float tile[32][33];
    const size_t base = (size_t)blockIdx.z * R * C;
    const int tx = threadIdx.x, ty = threadIdx.y;
    const int c0 = blockIdx.x * 32, r0 = blockIdx.y * 32;

#pragma unroll
    for (int j = 0; j < 32; j += 8)
        tile[ty + j][tx] = in[base + (size_t)(r0 + ty + j) * C + (c0 + tx)];
    __syncthreads();
#pragma unroll
    for (int j = 0; j < 32; j += 8)
        out[base + (size_t)(c0 + ty + j) * R + (r0 + tx)] = (half_t)tile[tx][ty + j];
}

// ---------------------------------------------------------------------------
// out[row] = scale * sum_i Xt[row*512 + i]   (row = b*D + e), wave per row.
// ---------------------------------------------------------------------------
__global__ __launch_bounds__(256)
void mean_rows(const half_t* __restrict__ Xt, float* __restrict__ out, float scale) {
    const int row  = blockIdx.x * 8 + (threadIdx.x >> 5);
    const int lane = threadIdx.x & 31;
    const half_t* p = Xt + (size_t)row * 512;
    float s = 0.0f;
#pragma unroll
    for (int it = 0; it < 2; ++it) {
        v8h v = *(const v8h*)(p + it * 256 + lane * 8);
#pragma unroll
        for (int r = 0; r < 8; ++r) s += (float)v[r];
    }
#pragma unroll
    for (int off = 16; off > 0; off >>= 1) s += __shfl_xor(s, off, 32);
    if (lane == 0) out[row] = s * scale;
}

// ---------------------------------------------------------------------------
// Workspace layout (f16):
//   adj : B*L*L = 33.6 MB | X0t : B*D*L = 33.6 MB | X1t : B*D*L = 33.6 MB
//   Y   : B*L*D = 33.6 MB | Wt  : NL*D*D = 1.0 MB     total ~135.3 MB
// ---------------------------------------------------------------------------
extern "C" void kernel_launch(void* const* d_in, const int* in_sizes, int n_in,
                              void* d_out, int out_size, void* d_ws, size_t ws_size,
                              hipStream_t stream) {
    const float* enc        = (const float*)d_in[0];   // (B,L,D) f32
    const float* ui_adj     = (const float*)d_in[1];   // (T,T)   f32
    const float* Ws         = (const float*)d_in[2];   // (NL,D,D) f32
    const int*   event_type = (const int*)d_in[3];     // (B,L)   int
    float*       out        = (float*)d_out;           // (B,D)   f32

    const size_t nBLL = (size_t)BB * LL * LL;
    const size_t nBDL = (size_t)BB * DD * LL;
    half_t* adj = (half_t*)d_ws;
    half_t* X0  = adj + nBLL;
    half_t* X1  = X0 + nBDL;
    half_t* Y   = X1 + nBDL;
    half_t* Wt  = Y  + nBDL;

    // Stage inputs into WMMA-friendly f16 layouts
    build_adj<<<dim3(LL / 16, BB), 256, 0, stream>>>(ui_adj, event_type, adj);
    transpose_cvt<<<dim3(DD / 32, LL / 32, BB), dim3(32, 8), 0, stream>>>(enc, X0, LL, DD);
    transpose_cvt<<<dim3(DD / 32, DD / 32, NLAYER), dim3(32, 8), 0, stream>>>(Ws, Wt, DD, DD);

    const dim3 gg(512 / 128, 512 / 128, BB);
    const unsigned long long sLL = (unsigned long long)LL * LL;
    const unsigned long long sDL = (unsigned long long)DD * LL;

    // Layer 0:  Y = adj @ X0   ;  X1 = relu(Y @ W0)  (stored transposed)
    gemm_wmma_f16<false, false><<<gg, 256, 0, stream>>>(adj, sLL, X0, sDL, Y, sDL);
    gemm_wmma_f16<true,  true ><<<gg, 256, 0, stream>>>(Y, sDL, Wt, 0ULL, X1, sDL);
    // Layer 1:  Y = adj @ X1   ;  X0 = relu(Y @ W1)  (stored transposed)
    gemm_wmma_f16<false, false><<<gg, 256, 0, stream>>>(adj, sLL, X1, sDL, Y, sDL);
    gemm_wmma_f16<true,  true ><<<gg, 256, 0, stream>>>(Y, sDL, Wt + (size_t)DD * DD, 0ULL, X0, sDL);

    // out[b,e] = mean_i X0t[b,e,i]
    mean_rows<<<(BB * DD) / 8, 256, 0, stream>>>(X0, out, 1.0f / (float)LL);
}
